// GraphSetConv_84490596647534
// MI455X (gfx1250) — matrix operand order repaired and checked by reading, not compiled, not run on hardware
//
#include <hip/hip_runtime.h>

// Problem constants (fixed by the reference)
#define NN    100000
#define EE    1600000
#define GG    2000
#define SS    40
#define MP    50
#define INF_  64
#define FD    128
#define NH    4
#define HDD   32
#define FFN_D 1024
#define RSET  (SS*MP)   // 2000 dense-batch rows

typedef __attribute__((ext_vector_type(16))) __bf16 v16bf;
typedef __attribute__((ext_vector_type(8)))  float  v8f;
typedef __attribute__((ext_vector_type(4)))  int    i32x4;
typedef __attribute__((ext_vector_type(8)))  int    i32x8;

// ---------- helpers ----------
__device__ __forceinline__ __bf16 f2bf(float f) {
  unsigned u = __builtin_bit_cast(unsigned, f);
  unsigned r = (u + 0x7FFFu + ((u >> 16) & 1u)) >> 16;   // round-to-nearest-even
  unsigned short s = (unsigned short)r;
  return __builtin_bit_cast(__bf16, s);
}

__device__ __forceinline__ float warp_sum(float s) {
#pragma unroll
  for (int o = 16; o > 0; o >>= 1) s += __shfl_xor(s, o, 32);
  return s;
}

// Issue a 1-D Tensor-Data-Mover load of `ndw` dwords from `gsrc` into LDS offset
// `lds_off` (descriptor per CDNA5 ISA 08_async_tensor §8; 2-group form for <=2D).
__device__ __forceinline__ void tdm_load_1d(const void* gsrc, unsigned lds_off, unsigned ndw) {
  unsigned long long ga = (unsigned long long)gsrc;
  i32x4 g0;
  g0[0] = 1;                                                  // count=1, user mode
  g0[1] = (int)lds_off;                                       // lds_addr
  g0[2] = (int)(unsigned)ga;                                  // global_addr[31:0]
  g0[3] = (int)(((unsigned)(ga >> 32) & 0x01FFFFFFu)          // global_addr[56:32]
                | 0x80000000u);                               // type=2 ("image")
  i32x8 g1;
  g1[0] = 2 << 16;                                            // data_size=4B, no multicast
  g1[1] = (int)((ndw & 0xFFFFu) << 16);                       // tensor_dim0[15:0]
  g1[2] = (int)((ndw >> 16) & 0xFFFFu);                       // tensor_dim0[31:16]
  g1[3] = (int)((ndw & 0xFFFFu) << 16);                       // tile_dim0 (ndw < 64K)
  g1[4] = 0;                                                  // tile_dim1/2 unused
  g1[5] = (int)ndw;                                           // tensor_dim0_stride
  g1[6] = 0;
  g1[7] = 0;
  asm volatile("tensor_load_to_lds %0, %1" : : "s"(g0), "s"(g1) : "memory");
}

// ---------- bf16 fragment packing (WMMA register layout, ISA 7.12.2) ----------
// Swizzled element = v16bf at index ((tile * kt + t) * 32 + lane).
// A: tile = 16-row group; lane<16 rows 0..15 K-pairs 0..7/16..23, lane>=16 K 8..15/24..31.
__global__ void k_pack_a(const float* __restrict__ A, v16bf* __restrict__ out, int M, int K) {
  int tid = blockIdx.x * blockDim.x + threadIdx.x;
  int kt = K >> 5;
  int total = (M >> 4) * kt * 32;
  if (tid >= total) return;
  int lane = tid & 31;
  int t = (tid >> 5) % kt;
  int tile = (tid >> 5) / kt;
  int m = lane & 15, half = lane >> 4;
  const float* ar = A + (size_t)(tile * 16 + m) * K + t * 32;
  v16bf v;
#pragma unroll
  for (int j = 0; j < 8; ++j) {
    int kk = (j < 4) ? (half * 8 + 2 * j) : (16 + half * 8 + 2 * (j - 4));
    v[2 * j]     = f2bf(ar[kk]);
    v[2 * j + 1] = f2bf(ar[kk + 1]);
  }
  out[tid] = v;
}
// B: tile = 16-col panel; lane<16 K 0..7/16..23, lane>=16 K 8..15/24..31; col = lane&15.
__global__ void k_pack_b(const float* __restrict__ B, v16bf* __restrict__ out, int K, int N) {
  int tid = blockIdx.x * blockDim.x + threadIdx.x;
  int kt = K >> 5;
  int total = (N >> 4) * kt * 32;
  if (tid >= total) return;
  int lane = tid & 31;
  int t = (tid >> 5) % kt;
  int p = (tid >> 5) / kt;
  int n = lane & 15, half = lane >> 4;
  v16bf v;
#pragma unroll
  for (int j = 0; j < 8; ++j) {
    int kk = t * 32 + ((j < 4) ? (half * 8 + 2 * j) : (16 + half * 8 + 2 * (j - 4)));
    v[2 * j]     = f2bf(B[(size_t)kk * N + p * 16 + n]);
    v[2 * j + 1] = f2bf(B[(size_t)(kk + 1) * N + p * 16 + n]);
  }
  out[tid] = v;
}
// gate A = [h | set_info] (K=256): same A swizzle, sources switched at k=128.
__global__ void k_pack_gate_a(const float* __restrict__ h, const float* __restrict__ zd2,
                              const int* __restrict__ gidx, v16bf* __restrict__ out) {
  int tid = blockIdx.x * blockDim.x + threadIdx.x;
  const int kt = 8;  // 256/32
  int total = (NN >> 4) * kt * 32;
  if (tid >= total) return;
  int lane = tid & 31;
  int t = (tid >> 5) % kt;
  int tile = (tid >> 5) / kt;
  int m = lane & 15, half = lane >> 4;
  int row = tile * 16 + m;
  const float* hrow  = h   + (size_t)row * FD;
  const float* sirow = zd2 + (size_t)gidx[row] * FD;
  v16bf v;
#pragma unroll
  for (int j = 0; j < 8; ++j) {
    int kk = t * 32 + ((j < 4) ? (half * 8 + 2 * j) : (16 + half * 8 + 2 * (j - 4)));
    const float* src = (kk < FD) ? (hrow + kk) : (sirow + kk - FD);
    v[2 * j]     = f2bf(src[0]);
    v[2 * j + 1] = f2bf(src[1]);
  }
  out[tid] = v;
}

// ---------- generic bf16 WMMA GEMM with TDM-staged A panel ----------
// grid = (M/16, N/128), block = 256 (8 waves, each owns a 16-col panel)
__global__ __launch_bounds__(256) void gemm_bf16_wmma(
    const v16bf* __restrict__ Asw, const v16bf* __restrict__ Bsw,
    const float* __restrict__ bias, float* __restrict__ C,
    int M, int Ncols, int K, int act) {
  __shared__ v16bf As[1024];              // 32 KB max panel (K<=1024), LDS offset 0
  int kt = K >> 5;
  int nfrag = kt * 32;
  const v16bf* Ablk = Asw + (size_t)blockIdx.x * nfrag;
  if (threadIdx.x == 0) {
    tdm_load_1d(Ablk, 0u, (unsigned)(nfrag * 8));   // panel bytes / 4
    __builtin_amdgcn_s_wait_tensorcnt(0);
  }
  __syncthreads();
  int wave = threadIdx.x >> 5, lane = threadIdx.x & 31;
  int colp = blockIdx.y * 8 + wave;
  const v16bf* Bblk = Bsw + ((size_t)colp * kt) * 32 + lane;
  v8f acc = {};
  for (int t = 0; t < kt; ++t) {
    if (t + 1 < kt) __builtin_prefetch(Bblk + (size_t)(t + 1) * 32, 0, 1);
    v16bf a = As[t * 32 + lane];
    v16bf b = Bblk[(size_t)t * 32];
    acc = __builtin_amdgcn_wmma_f32_16x16x32_bf16(false, a, false, b, (short)0, acc,
                                                  false, false);
  }
  int half = lane >> 4, n = lane & 15;
  int col = colp * 16 + n;
  float bv = bias ? bias[col] : 0.0f;
#pragma unroll
  for (int r = 0; r < 8; ++r) {
    int row = blockIdx.x * 16 + r + 8 * half;
    float v = acc[r] + bv;
    if (act == 1) v = fmaxf(v, 0.0f);
    C[(size_t)row * Ncols + col] = v;
  }
}

// ---------- gate GEMM (K=256) + fused sigmoid blend epilogue ----------
__global__ __launch_bounds__(256) void gate_fuse_wmma(
    const v16bf* __restrict__ Asw, const v16bf* __restrict__ Bsw,
    const float* __restrict__ bg, const float* __restrict__ h,
    const float* __restrict__ zd2, const int* __restrict__ gidx,
    float* __restrict__ out) {
  const int kt = 8;
  __shared__ v16bf As[kt * 32];           // 8 KB
  const v16bf* Ablk = Asw + (size_t)blockIdx.x * (kt * 32);
  if (threadIdx.x == 0) {
    tdm_load_1d(Ablk, 0u, (unsigned)(kt * 32 * 8));
    __builtin_amdgcn_s_wait_tensorcnt(0);
  }
  __syncthreads();
  int wave = threadIdx.x >> 5, lane = threadIdx.x & 31;
  const v16bf* Bblk = Bsw + ((size_t)wave * kt) * 32 + lane;
  v8f acc = {};
#pragma unroll
  for (int t = 0; t < kt; ++t) {
    v16bf a = As[t * 32 + lane];
    v16bf b = Bblk[(size_t)t * 32];
    acc = __builtin_amdgcn_wmma_f32_16x16x32_bf16(false, a, false, b, (short)0, acc,
                                                  false, false);
  }
  int half = lane >> 4, n = lane & 15;
  int col = wave * 16 + n;
  float bgv = bg[col];
#pragma unroll
  for (int r = 0; r < 8; ++r) {
    int row = blockIdx.x * 16 + r + 8 * half;
    float g = 1.0f / (1.0f + __expf(-(acc[r] + bgv)));
    float hv = h[(size_t)row * FD + col];
    float sv = zd2[(size_t)gidx[row] * FD + col];
    out[(size_t)row * FD + col] = g * sv + (1.0f - g) * hv;
  }
}

// ---------- elementwise / graph kernels ----------
__global__ void k_fill_f(float* p, float v, int n) {
  int i = blockIdx.x * blockDim.x + threadIdx.x;
  if (i < n) p[i] = v;
}
__global__ void k_fill_i(int* p, int v, int n) {
  int i = blockIdx.x * blockDim.x + threadIdx.x;
  if (i < n) p[i] = v;
}
__global__ void k_deg(const int* __restrict__ col, float* __restrict__ deg, int E) {
  int e = blockIdx.x * blockDim.x + threadIdx.x;
  if (e < E) atomicAdd(&deg[col[e]], 1.0f);
}
__global__ void k_rsqrt_inplace(float* d, int n) {
  int i = blockIdx.x * blockDim.x + threadIdx.x;
  if (i < n) d[i] = rsqrtf(d[i]);
}
// one wave per edge: gather xw[row] (float4/lane), scale, atomic scatter into agg[col]
__global__ __launch_bounds__(256) void k_edge(
    const int* __restrict__ row, const int* __restrict__ col,
    const float* __restrict__ dinv, const float* __restrict__ xw,
    float* __restrict__ agg, int E) {
  int tid = blockIdx.x * blockDim.x + threadIdx.x;
  int e = tid >> 5, lane = tid & 31;
  if (e >= E) return;
  int r = row[e], c = col[e];
  float w = dinv[r] * dinv[c];
  const float4* src = (const float4*)(xw + (size_t)r * FD);
  float4 v = src[lane];
  float* dst = agg + (size_t)c * FD + lane * 4;
  atomicAdd(dst + 0, w * v.x);
  atomicAdd(dst + 1, w * v.y);
  atomicAdd(dst + 2, w * v.z);
  atomicAdd(dst + 3, w * v.w);
}
// h = relu((agg + 2*dinv^2*xw + b_gcn) * rsqrt(1+eps) * bn_g + bn_b)   (in place on agg)
__global__ void k_hcombine(float* __restrict__ agg, const float* __restrict__ xw,
                           const float* __restrict__ dinv, const float* __restrict__ b_gcn,
                           const float* __restrict__ bn_g, const float* __restrict__ bn_b) {
  int tid = blockIdx.x * blockDim.x + threadIdx.x;
  if (tid >= NN * FD) return;
  int n = tid / FD, f = tid - n * FD;
  float di = dinv[n];
  float val = agg[tid] + 2.0f * di * di * xw[tid] + b_gcn[f];
  val = val * rsqrtf(1.0f + 1e-5f) * bn_g[f] + bn_b[f];
  agg[tid] = fmaxf(val, 0.0f);
}
__global__ void k_pool_cnt(const int* __restrict__ batch, float* __restrict__ cntg) {
  int n = blockIdx.x * blockDim.x + threadIdx.x;
  if (n < NN) atomicAdd(&cntg[batch[n]], 1.0f);
}
__global__ void k_pool_acc(const int* __restrict__ batch, const float* __restrict__ h,
                           float* __restrict__ z) {
  int tid = blockIdx.x * blockDim.x + threadIdx.x;
  if (tid >= NN * FD) return;
  int n = tid / FD, f = tid - n * FD;
  atomicAdd(&z[(size_t)batch[n] * FD + f], h[tid]);
}
__global__ void k_pool_div(float* __restrict__ z, const float* __restrict__ cntg) {
  int tid = blockIdx.x * blockDim.x + threadIdx.x;
  if (tid < GG * FD) z[tid] /= cntg[tid / FD];
}
__global__ void k_cnt_s(const int* __restrict__ sb, int* __restrict__ cnt_s) {
  int g = blockIdx.x * blockDim.x + threadIdx.x;
  if (g < GG) atomicAdd(&cnt_s[sb[g]], 1);
}
__global__ void k_pos(const int* __restrict__ sb, const int* __restrict__ cnt_s,
                      int* __restrict__ idx) {
  __shared__ int st[SS];
  if (threadIdx.x == 0) {
    int a = 0;
    for (int i = 0; i < SS; ++i) { st[i] = a; a += cnt_s[i]; }
  }
  __syncthreads();
  for (int g = threadIdx.x; g < GG; g += blockDim.x) {
    int p = g - st[sb[g]];
    idx[g] = sb[g] * MP + p;
  }
}
__global__ void k_scatter_zd(const float* __restrict__ z, const int* __restrict__ idx,
                             float* __restrict__ zd) {
  int tid = blockIdx.x * blockDim.x + threadIdx.x;
  if (tid >= GG * FD) return;
  int g = tid / FD, f = tid - g * FD;
  zd[(size_t)idx[g] * FD + f] = z[tid];
}
__global__ void k_gidx(const int* __restrict__ batch, const int* __restrict__ idx,
                       int* __restrict__ gidx) {
  int n = blockIdx.x * blockDim.x + threadIdx.x;
  if (n < NN) gidx[n] = idx[batch[n]];
}

// ---------- LayerNorm over 128 features; optional residual ----------
__global__ __launch_bounds__(128) void k_ln(const float* __restrict__ x,
                                            const float* __restrict__ resid,
                                            const float* __restrict__ g,
                                            const float* __restrict__ b,
                                            float* __restrict__ out, int rows) {
  __shared__ float sh[4];
  int row = blockIdx.x;
  if (row >= rows) return;
  int t = threadIdx.x;
  size_t base = (size_t)row * FD;
  float v = x[base + t];
  if (resid) v += resid[base + t];
  float s = warp_sum(v);
  if ((t & 31) == 0) sh[t >> 5] = s;
  __syncthreads();
  float mean = (sh[0] + sh[1] + sh[2] + sh[3]) * (1.0f / FD);
  __syncthreads();
  float d = v - mean;
  float s2 = warp_sum(d * d);
  if ((t & 31) == 0) sh[t >> 5] = s2;
  __syncthreads();
  float var = (sh[0] + sh[1] + sh[2] + sh[3]) * (1.0f / FD);
  out[base + t] = d * rsqrtf(var + 1e-5f) * g[t] + b[t];
}

// ---------- attention (tiny: 40x4 heads of 50x50x32) ----------
__global__ void k_scores(const float* __restrict__ q, const float* __restrict__ k,
                         float* __restrict__ sc) {
  int tid = blockIdx.x * blockDim.x + threadIdx.x;
  if (tid >= SS * NH * MP * MP) return;
  int kp = tid % MP, t2 = tid / MP;
  int qp = t2 % MP, t3 = t2 / MP;
  int hh = t3 % NH, s = t3 / NH;
  const float* qr = q + ((size_t)(s * MP + qp) * FD + hh * HDD);
  const float* kr = k + ((size_t)(s * MP + kp) * FD + hh * HDD);
  float d = 0.0f;
#pragma unroll
  for (int i = 0; i < HDD; ++i) d += qr[i] * kr[i];
  sc[tid] = d * 0.17677669529663687f;  // 1/sqrt(32)
}
__global__ void k_softmax(float* __restrict__ sc, const int* __restrict__ cnt_s) {
  int tid = blockIdx.x * blockDim.x + threadIdx.x;
  if (tid >= SS * NH * MP) return;
  int s = tid / (NH * MP);
  int valid = cnt_s[s];
  float* r = sc + (size_t)tid * MP;
  float mx = -1e30f;
  for (int kp = 0; kp < MP; ++kp) {
    float v = (kp < valid) ? r[kp] : -1e9f;
    mx = fmaxf(mx, v);
  }
  float sum = 0.0f;
  for (int kp = 0; kp < MP; ++kp) {
    float v = (kp < valid) ? r[kp] : -1e9f;
    float e = __expf(v - mx);
    r[kp] = e;
    sum += e;
  }
  float inv = 1.0f / sum;
  for (int kp = 0; kp < MP; ++kp) r[kp] *= inv;
}
__global__ void k_apply(const float* __restrict__ sc, const float* __restrict__ v,
                        float* __restrict__ ob) {
  int tid = blockIdx.x * blockDim.x + threadIdx.x;
  if (tid >= SS * MP * NH * HDD) return;
  int d = tid % HDD, t2 = tid / HDD;
  int hh = t2 % NH, t3 = t2 / NH;
  int qp = t3 % MP, s = t3 / MP;
  const float* arow = sc + (size_t)((s * NH + hh) * MP + qp) * MP;
  int col = hh * HDD + d;
  float acc = 0.0f;
  for (int kp = 0; kp < MP; ++kp)
    acc += arow[kp] * v[(size_t)(s * MP + kp) * FD + col];
  ob[(size_t)(s * MP + qp) * FD + col] = acc;
}

// ---------- launcher ----------
extern "C" void kernel_launch(void* const* d_in, const int* in_sizes, int n_in,
                              void* d_out, int out_size, void* d_ws, size_t ws_size,
                              hipStream_t stream) {
  const float* x     = (const float*)d_in[0];
  const int*   eidx  = (const int*)d_in[1];
  const int*   batch = (const int*)d_in[2];
  const int*   sb    = (const int*)d_in[3];
  const float* W_gcn = (const float*)d_in[4];
  const float* b_gcn = (const float*)d_in[5];
  const float* bn_g  = (const float*)d_in[6];
  const float* bn_b  = (const float*)d_in[7];
  const float* lnp_g = (const float*)d_in[8];
  const float* lnp_b = (const float*)d_in[9];
  const float* Wq = (const float*)d_in[10]; const float* bq = (const float*)d_in[11];
  const float* Wk = (const float*)d_in[12]; const float* bk = (const float*)d_in[13];
  const float* Wv = (const float*)d_in[14]; const float* bv = (const float*)d_in[15];
  const float* Wo = (const float*)d_in[16]; const float* bo = (const float*)d_in[17];
  const float* l1g = (const float*)d_in[18]; const float* l1b = (const float*)d_in[19];
  const float* W1 = (const float*)d_in[20]; const float* b1 = (const float*)d_in[21];
  const float* W2 = (const float*)d_in[22]; const float* b2 = (const float*)d_in[23];
  const float* l2g = (const float*)d_in[24]; const float* l2b = (const float*)d_in[25];
  const float* Wg = (const float*)d_in[26]; const float* bg = (const float*)d_in[27];
  float* out = (float*)d_out;
  const int* erow = eidx;
  const int* ecol = eidx + EE;

  char* w = (char*)d_ws;
  auto carve = [&](size_t nbytes) {
    char* p = w;
    w += (nbytes + 255) & ~(size_t)255;
    return p;
  };
  // fp32 buffers
  float* xw   = (float*)carve(sizeof(float) * (size_t)NN * FD);
  float* hbuf = (float*)carve(sizeof(float) * (size_t)NN * FD);   // agg -> h
  float* deg  = (float*)carve(sizeof(float) * NN);                // deg -> dinv
  float* z    = (float*)carve(sizeof(float) * GG * FD);
  float* cntg = (float*)carve(sizeof(float) * GG);
  float* zd   = (float*)carve(sizeof(float) * RSET * FD);
  float* zn   = (float*)carve(sizeof(float) * RSET * FD);
  float* qb   = (float*)carve(sizeof(float) * RSET * FD);
  float* kb   = (float*)carve(sizeof(float) * RSET * FD);
  float* vb   = (float*)carve(sizeof(float) * RSET * FD);
  float* sc   = (float*)carve(sizeof(float) * SS * NH * MP * MP);
  float* ob   = (float*)carve(sizeof(float) * RSET * FD);
  float* op   = (float*)carve(sizeof(float) * RSET * FD);
  float* zd1  = (float*)carve(sizeof(float) * RSET * FD);
  float* ffh  = (float*)carve(sizeof(float) * (size_t)RSET * FFN_D);
  float* ff2  = (float*)carve(sizeof(float) * RSET * FD);
  float* zd2  = (float*)carve(sizeof(float) * RSET * FD);
  int*   cnts = (int*)carve(sizeof(int) * SS);
  int*   idx  = (int*)carve(sizeof(int) * GG);
  int*   gidx = (int*)carve(sizeof(int) * NN);
  // packed bf16 fragment buffers: (M/16)*(K/32)*32 v16bf elements (32B each)
  auto afrags = [](long M, long K) { return (size_t)(M / 16) * (K / 32) * 32; };
  v16bf* xa    = (v16bf*)carve(32 * afrags(NN, INF_));
  v16bf* zna   = (v16bf*)carve(32 * afrags(RSET, FD));
  v16bf* oba   = (v16bf*)carve(32 * afrags(RSET, FD));
  v16bf* zd1a  = (v16bf*)carve(32 * afrags(RSET, FD));
  v16bf* ffha  = (v16bf*)carve(32 * afrags(RSET, FFN_D));
  v16bf* gatea = (v16bf*)carve(32 * afrags(NN, 2 * FD));
  v16bf* Wgcnp = (v16bf*)carve(32 * afrags(FD, INF_));
  v16bf* Wqp   = (v16bf*)carve(32 * afrags(FD, FD));
  v16bf* Wkp   = (v16bf*)carve(32 * afrags(FD, FD));
  v16bf* Wvp   = (v16bf*)carve(32 * afrags(FD, FD));
  v16bf* Wop   = (v16bf*)carve(32 * afrags(FD, FD));
  v16bf* W1p   = (v16bf*)carve(32 * afrags(FFN_D, FD));
  v16bf* W2p   = (v16bf*)carve(32 * afrags(FD, FFN_D));
  v16bf* Wgp   = (v16bf*)carve(32 * afrags(FD, 2 * FD));

  const int B = 256;
  auto nb = [](long n, int b) { return (int)((n + b - 1) / b); };

  // init
  k_fill_f<<<nb(NN, B), B, 0, stream>>>(deg, 2.0f, NN);                 // self-loop weight 2
  k_fill_f<<<nb((long)NN * FD, B), B, 0, stream>>>(hbuf, 0.0f, NN * FD);
  k_fill_f<<<nb(GG * FD, B), B, 0, stream>>>(z, 0.0f, GG * FD);
  k_fill_f<<<nb(GG, B), B, 0, stream>>>(cntg, 0.0f, GG);
  k_fill_f<<<nb(RSET * FD, B), B, 0, stream>>>(zd, 0.0f, RSET * FD);
  k_fill_i<<<1, SS, 0, stream>>>(cnts, 0, SS);

  // pack weights (tiny, once per launch)
  k_pack_b<<<nb((long)afrags(FD, INF_), B), B, 0, stream>>>(W_gcn, Wgcnp, INF_, FD);
  k_pack_b<<<nb((long)afrags(FD, FD), B), B, 0, stream>>>(Wq, Wqp, FD, FD);
  k_pack_b<<<nb((long)afrags(FD, FD), B), B, 0, stream>>>(Wk, Wkp, FD, FD);
  k_pack_b<<<nb((long)afrags(FD, FD), B), B, 0, stream>>>(Wv, Wvp, FD, FD);
  k_pack_b<<<nb((long)afrags(FD, FD), B), B, 0, stream>>>(Wo, Wop, FD, FD);
  k_pack_b<<<nb((long)afrags(FFN_D, FD), B), B, 0, stream>>>(W1, W1p, FD, FFN_D);
  k_pack_b<<<nb((long)afrags(FD, FFN_D), B), B, 0, stream>>>(W2, W2p, FFN_D, FD);
  k_pack_b<<<nb((long)afrags(FD, 2 * FD), B), B, 0, stream>>>(Wg, Wgp, 2 * FD, FD);

  // GCN: xw = x @ W_gcn (WMMA, TDM-staged A)
  k_pack_a<<<nb((long)afrags(NN, INF_), B), B, 0, stream>>>(x, xa, NN, INF_);
  gemm_bf16_wmma<<<dim3(NN / 16, 1), B, 0, stream>>>(xa, Wgcnp, nullptr, xw, NN, FD, INF_, 0);
  // degree + norm
  k_deg<<<nb(EE, B), B, 0, stream>>>(ecol, deg, EE);
  k_rsqrt_inplace<<<nb(NN, B), B, 0, stream>>>(deg, NN);  // deg -> dinv
  // edge gather/scatter (memory-bound heavy hitter: ~1.6 GB)
  k_edge<<<nb((long)EE * 32, B), B, 0, stream>>>(erow, ecol, deg, xw, hbuf, EE);
  // h = BN(ReLU-fused) combine
  k_hcombine<<<nb((long)NN * FD, B), B, 0, stream>>>(hbuf, xw, deg, b_gcn, bn_g, bn_b);
  // mean pool
  k_pool_cnt<<<nb(NN, B), B, 0, stream>>>(batch, cntg);
  k_pool_acc<<<nb((long)NN * FD, B), B, 0, stream>>>(batch, hbuf, z);
  k_pool_div<<<nb(GG * FD, B), B, 0, stream>>>(z, cntg);
  // dense batch
  k_cnt_s<<<nb(GG, B), B, 0, stream>>>(sb, cnts);
  k_pos<<<1, B, 0, stream>>>(sb, cnts, idx);
  k_scatter_zd<<<nb(GG * FD, B), B, 0, stream>>>(z, idx, zd);
  k_gidx<<<nb(NN, B), B, 0, stream>>>(batch, idx, gidx);
  // pre-LN MHA
  k_ln<<<RSET, FD, 0, stream>>>(zd, nullptr, lnp_g, lnp_b, zn, RSET);
  k_pack_a<<<nb((long)afrags(RSET, FD), B), B, 0, stream>>>(zn, zna, RSET, FD);
  gemm_bf16_wmma<<<dim3(RSET / 16, 1), B, 0, stream>>>(zna, Wqp, bq, qb, RSET, FD, FD, 0);
  gemm_bf16_wmma<<<dim3(RSET / 16, 1), B, 0, stream>>>(zna, Wkp, bk, kb, RSET, FD, FD, 0);
  gemm_bf16_wmma<<<dim3(RSET / 16, 1), B, 0, stream>>>(zna, Wvp, bv, vb, RSET, FD, FD, 0);
  k_scores<<<nb((long)SS * NH * MP * MP, B), B, 0, stream>>>(qb, kb, sc);
  k_softmax<<<nb(SS * NH * MP, B), B, 0, stream>>>(sc, cnts);
  k_apply<<<nb((long)SS * MP * NH * HDD, B), B, 0, stream>>>(sc, vb, ob);
  k_pack_a<<<nb((long)afrags(RSET, FD), B), B, 0, stream>>>(ob, oba, RSET, FD);
  gemm_bf16_wmma<<<dim3(RSET / 16, 1), B, 0, stream>>>(oba, Wop, bo, op, RSET, FD, FD, 0);
  k_ln<<<RSET, FD, 0, stream>>>(zd, op, l1g, l1b, zd1, RSET);
  // FFN
  k_pack_a<<<nb((long)afrags(RSET, FD), B), B, 0, stream>>>(zd1, zd1a, RSET, FD);
  gemm_bf16_wmma<<<dim3(RSET / 16, FFN_D / 128), B, 0, stream>>>(zd1a, W1p, b1, ffh, RSET, FFN_D, FD, 1);
  k_pack_a<<<nb((long)afrags(RSET, FFN_D), B), B, 0, stream>>>(ffh, ffha, RSET, FFN_D);
  gemm_bf16_wmma<<<dim3(RSET / 16, 1), B, 0, stream>>>(ffha, W2p, b2, ff2, RSET, FD, FFN_D, 0);
  k_ln<<<RSET, FD, 0, stream>>>(zd1, ff2, l2g, l2b, zd2, RSET);
  // gated fusion (packed concat-A + TDM-staged WMMA + sigmoid blend) -> final output
  k_pack_gate_a<<<nb((long)afrags(NN, 2 * FD), B), B, 0, stream>>>(hbuf, zd2, gidx, gatea);
  gate_fuse_wmma<<<dim3(NN / 16, 1), B, 0, stream>>>(gatea, Wgp, bg, hbuf, zd2, gidx, out);

  (void)in_sizes; (void)n_in; (void)out_size; (void)ws_size;
}
